// BasicBlock_6339371729554
// MI455X (gfx1250) — compile-verified, matrix operand order checked by reading, not run on hardware
//
#include <hip/hip_runtime.h>
#include <hip/hip_bf16.h>
#include <math.h>

// ---------------- problem constants ----------------
#define Bn 4
#define Hn 56
#define Wn 56
#define Cn 256
#define Gn 16
#define Kn 9
#define GCn 16
#define Mrows (Bn * Hn * Wn)   // 12544

typedef __hip_bfloat16 bf16_t;
typedef __attribute__((ext_vector_type(16))) __bf16 v16bf;
typedef __attribute__((ext_vector_type(8)))  float  v8f;
typedef __attribute__((ext_vector_type(4)))  unsigned int uint4v;

__device__ __forceinline__ float gelu_exact(float v) {
    return 0.5f * v * (1.0f + erff(v * 0.70710678118654752440f));
}

// Load one 16-lane WMMA fragment row: lane holds 16 bf16 of row `rowptr`,
// K-pattern per CDNA5 ISA 16-bit A 16x32 layout:
//   elements 0..7  -> K = k0 + half*8 .. +7
//   elements 8..15 -> K = k0 + 16 + half*8 .. +7
__device__ __forceinline__ v16bf load_frag(const bf16_t* __restrict__ rowptr,
                                           int k0, int half) {
    union { uint4v u[2]; v16bf v; } t;
    t.u[0] = *reinterpret_cast<const uint4v*>(rowptr + k0 + half * 8);
    t.u[1] = *reinterpret_cast<const uint4v*>(rowptr + k0 + 16 + half * 8);
    return t.v;
}

// ---------------- register-tiled WMMA GEMM ----------------------------------
// out[M,N] = A[M,K] @ Bt[N,K]^T + bias ; bf16 inputs, f32 accumulate.
// Each wave computes an MT x NT grid of 16x16 tiles: per 32-wide K step it
// loads MT + NT fragments and issues MT*NT v_wmma_f32_16x16x32_bf16.
// act: 0 = none, 1 = exact GELU. outF / outB may be null.
template <int MT, int NT>
__global__ void gemm_bf16_wmma(const bf16_t* __restrict__ A,
                               const bf16_t* __restrict__ Bt,
                               const float*  __restrict__ bias,
                               float* __restrict__ outF,
                               bf16_t* __restrict__ outB,
                               int M, int N, int K, int act, int totalWaves)
{
    const int lane = threadIdx.x & 31;
    const int wave = (blockIdx.x * blockDim.x + threadIdx.x) >> 5;
    if (wave >= totalWaves) return;               // whole-wave exit, EXEC stays all-1

    const int ntn   = N / (16 * NT);              // wave-tiles across N
    const int tileM = (wave / ntn) * (16 * MT);
    const int tileN = (wave % ntn) * (16 * NT);
    const int r     = lane & 15;
    const int half  = lane >> 4;

    const bf16_t* arow[MT];
    const bf16_t* brow[NT];
#pragma unroll
    for (int i = 0; i < MT; ++i) arow[i] = A  + (size_t)(tileM + i * 16 + r) * K;
#pragma unroll
    for (int j = 0; j < NT; ++j) brow[j] = Bt + (size_t)(tileN + j * 16 + r) * K;

    v8f acc[MT][NT] = {};

    for (int k0 = 0; k0 < K; k0 += 32) {
        if (k0 + 32 < K) {                        // near-scope prefetch of next slab
#pragma unroll
            for (int i = 0; i < MT; ++i) __builtin_prefetch(arow[i] + k0 + 32, 0, 3);
#pragma unroll
            for (int j = 0; j < NT; ++j) __builtin_prefetch(brow[j] + k0 + 32, 0, 3);
        }
        v16bf af[MT], bfr[NT];
#pragma unroll
        for (int i = 0; i < MT; ++i) af[i] = load_frag(arow[i], k0, half);
#pragma unroll
        for (int j = 0; j < NT; ++j) bfr[j] = load_frag(brow[j], k0, half);
#pragma unroll
        for (int i = 0; i < MT; ++i)
#pragma unroll
            for (int j = 0; j < NT; ++j)
                acc[i][j] = __builtin_amdgcn_wmma_f32_16x16x32_bf16(
                    /*neg_a=*/false, af[i], /*neg_b=*/false, bfr[j],
                    /*c_mod=*/(short)0, acc[i][j],
                    /*reuse_a=*/false, /*reuse_b=*/false);
    }

#pragma unroll
    for (int i = 0; i < MT; ++i) {
#pragma unroll
        for (int j = 0; j < NT; ++j) {
            const int col = tileN + j * 16 + r;
            const float bv = bias ? bias[col] : 0.0f;
#pragma unroll
            for (int v = 0; v < 8; ++v) {
                const int row = tileM + i * 16 + v + 8 * half; // C/D VGPR layout
                float val = acc[i][j][v] + bv;
                if (act == 1) val = gelu_exact(val);
                const size_t idx = (size_t)row * N + col;
                if (outF) outF[idx] = val;
                if (outB) outB[idx] = __float2bfloat16(val);
            }
        }
    }
}

// ---------------- weight transpose + f32 -> bf16:  Wt[n][k] = W[k][n] --------
__global__ void transpose_w_bf16(const float* __restrict__ W,
                                 bf16_t* __restrict__ Wt, int K, int N)
{
    int i = blockIdx.x * blockDim.x + threadIdx.x;
    if (i >= K * N) return;
    int n = i / K, k = i - n * K;
    Wt[i] = __float2bfloat16(W[(size_t)k * N + n]);
}

// ---------------- plain f32 -> bf16 copy ------------------------------------
__global__ void cvt_bf16(const float* __restrict__ in, bf16_t* __restrict__ out, int n)
{
    int i = blockIdx.x * blockDim.x + threadIdx.x;
    if (i < n) out[i] = __float2bfloat16(in[i]);
}

// ---------------- depthwise 3x3 conv + LN + GELU (block = one pixel) --------
__global__ void dwconv_ln_gelu(const float* __restrict__ x,
                               const float* __restrict__ wdw,
                               const float* __restrict__ bdw,
                               const float* __restrict__ g,
                               const float* __restrict__ bt,
                               float* __restrict__ outF,
                               bf16_t* __restrict__ outB)
{
    const int pix = blockIdx.x;            // 0 .. B*H*W-1
    const int c   = threadIdx.x;           // 0 .. 255
    const int b   = pix / (Hn * Wn);
    const int hw  = pix - b * (Hn * Wn);
    const int h   = hw / Wn, w = hw - h * Wn;

    float s = bdw[c];
#pragma unroll
    for (int ky = 0; ky < 3; ++ky) {
#pragma unroll
        for (int kx = 0; kx < 3; ++kx) {
            int ih = h + ky - 1, iw = w + kx - 1;
            if (ih >= 0 && ih < Hn && iw >= 0 && iw < Wn)
                s += x[((size_t)(b * Hn * Wn + ih * Wn + iw)) * Cn + c] *
                     wdw[(ky * 3 + kx) * Cn + c];
        }
    }
    __shared__ float red[Cn];
    red[c] = s;
    __syncthreads();
    for (int o = Cn / 2; o > 0; o >>= 1) {
        if (c < o) red[c] += red[c + o];
        __syncthreads();
    }
    const float mu = red[0] * (1.0f / Cn);
    __syncthreads();
    const float d = s - mu;
    red[c] = d * d;
    __syncthreads();
    for (int o = Cn / 2; o > 0; o >>= 1) {
        if (c < o) red[c] += red[c + o];
        __syncthreads();
    }
    const float var = red[0] * (1.0f / Cn);
    float y = d * rsqrtf(var + 1e-5f) * g[c] + bt[c];
    y = gelu_exact(y);
    const size_t idx = (size_t)pix * Cn + c;
    outF[idx] = y;
    outB[idx] = __float2bfloat16(y);
}

// ---------------- DCNv3 core: softmax(mask) + bilinear gather ---------------
// one thread per (pixel, group, channel)
__global__ void dcn_core(const float* __restrict__ xp,
                         const float* __restrict__ off,
                         const float* __restrict__ mlog,
                         bf16_t* __restrict__ outB)
{
    const int t = blockIdx.x * blockDim.x + threadIdx.x; // B*HW*G*GC
    const int c     = t & (GCn - 1);
    const int pg    = t >> 4;
    const int g     = pg & (Gn - 1);
    const int pixel = pg >> 4;
    const int b  = pixel / (Hn * Wn);
    const int hw = pixel - b * (Hn * Wn);
    const int h  = hw / Wn, w = hw - h * Wn;

    const float* __restrict__ ob = off  + (size_t)pg * (Kn * 2);
    const float* __restrict__ ml = mlog + (size_t)pg * Kn;

    // softmax over K = 9 in registers
    float lg[Kn], lm = -3.0e38f;
#pragma unroll
    for (int k = 0; k < Kn; ++k) { lg[k] = ml[k]; lm = fmaxf(lm, lg[k]); }
    float ssum = 0.0f;
#pragma unroll
    for (int k = 0; k < Kn; ++k) { lg[k] = __expf(lg[k] - lm); ssum += lg[k]; }
    const float inv = 1.0f / ssum;

    const float* __restrict__ xbase =
        xp + (size_t)b * (Hn * Wn) * Cn + g * GCn + c;

    float acc = 0.0f;
#pragma unroll
    for (int k = 0; k < Kn; ++k) {
        const float px = (float)w + (float)(k % 3) - 1.0f + ob[2 * k];
        const float py = (float)h + (float)(k / 3) - 1.0f + ob[2 * k + 1];
        const float x0 = floorf(px), y0 = floorf(py);
        const float fx = px - x0, fy = py - y0;
        const int ix0 = (int)x0, iy0 = (int)y0;
        float bi = 0.0f;
#pragma unroll
        for (int jj = 0; jj < 2; ++jj) {
#pragma unroll
            for (int ii = 0; ii < 2; ++ii) {
                const int cx = ix0 + ii, cy = iy0 + jj;
                const float wgt = (ii ? fx : 1.0f - fx) * (jj ? fy : 1.0f - fy);
                if (cx >= 0 && cx < Wn && cy >= 0 && cy < Hn)
                    bi += wgt * xbase[(size_t)(cy * Wn + cx) * Cn];
            }
        }
        acc += lg[k] * inv * bi;
    }
    outB[(size_t)pixel * Cn + g * GCn + c] = __float2bfloat16(acc);
}

// ---------------- residual + gamma * LN (block = one row) -------------------
__global__ void residual_ln(const float* __restrict__ xin,
                            const float* __restrict__ y,
                            const float* __restrict__ g,
                            const float* __restrict__ bt,
                            const float* __restrict__ gamma,
                            float* __restrict__ outF,
                            bf16_t* __restrict__ outB)
{
    const int row = blockIdx.x;
    const int c   = threadIdx.x;
    const size_t idx = (size_t)row * Cn + c;
    const float v = y[idx];

    __shared__ float red[Cn];
    red[c] = v;
    __syncthreads();
    for (int o = Cn / 2; o > 0; o >>= 1) {
        if (c < o) red[c] += red[c + o];
        __syncthreads();
    }
    const float mu = red[0] * (1.0f / Cn);
    __syncthreads();
    const float d = v - mu;
    red[c] = d * d;
    __syncthreads();
    for (int o = Cn / 2; o > 0; o >>= 1) {
        if (c < o) red[c] += red[c + o];
        __syncthreads();
    }
    const float var = red[0] * (1.0f / Cn);
    const float ln  = d * rsqrtf(var + 1e-5f) * g[c] + bt[c];
    const float out = xin[idx] + gamma[c] * ln;
    if (outF) outF[idx] = out;
    if (outB) outB[idx] = __float2bfloat16(out);
}

// ---------------- host orchestration ----------------------------------------
template <int MT, int NT>
static inline void launch_gemm(const bf16_t* A, const bf16_t* Bt, const float* bias,
                               float* outF, bf16_t* outB,
                               int M, int N, int K, int act, hipStream_t s)
{
    const int waves  = (M / (16 * MT)) * (N / (16 * NT));
    const int blocks = (waves + 7) / 8;           // 8 waves (256 thr) per block
    gemm_bf16_wmma<MT, NT><<<blocks, 256, 0, s>>>(A, Bt, bias, outF, outB,
                                                  M, N, K, act, waves);
}

extern "C" void kernel_launch(void* const* d_in, const int* in_sizes, int n_in,
                              void* d_out, int out_size, void* d_ws, size_t ws_size,
                              hipStream_t stream)
{
    const float* x      = (const float*)d_in[0];
    const float* w_in   = (const float*)d_in[1];
    const float* b_in   = (const float*)d_in[2];
    const float* w_dw   = (const float*)d_in[3];
    const float* b_dw   = (const float*)d_in[4];
    const float* lndw_g = (const float*)d_in[5];
    const float* lndw_b = (const float*)d_in[6];
    const float* w_off  = (const float*)d_in[7];
    const float* b_off  = (const float*)d_in[8];
    const float* w_mask = (const float*)d_in[9];
    const float* b_mask = (const float*)d_in[10];
    const float* w_out  = (const float*)d_in[11];
    const float* b_out  = (const float*)d_in[12];
    const float* ln1_g  = (const float*)d_in[13];
    const float* ln1_b  = (const float*)d_in[14];
    const float* w_fc1  = (const float*)d_in[15];
    const float* b_fc1  = (const float*)d_in[16];
    const float* w_fc2  = (const float*)d_in[17];
    const float* b_fc2  = (const float*)d_in[18];
    const float* ln2_g  = (const float*)d_in[19];
    const float* ln2_b  = (const float*)d_in[20];
    const float* gamma1 = (const float*)d_in[21];
    const float* gamma2 = (const float*)d_in[22];

    char*  ws  = (char*)d_ws;
    size_t pos = 0;
    auto alloc = [&](size_t bytes) -> void* {
        void* p = ws + pos;
        pos = (pos + bytes + 255) & ~(size_t)255;
        return p;
    };
    const size_t actF = (size_t)Mrows * Cn * sizeof(float);
    const size_t actB = (size_t)Mrows * Cn * sizeof(bf16_t);

    bf16_t* x_bf    = (bf16_t*)alloc(actB);
    float*  x1_f    = (float*) alloc(actF);
    bf16_t* x1_bf   = (bf16_t*)alloc(actB);
    float*  xp_f    = (float*) alloc(actF);
    float*  offbuf  = (float*) alloc((size_t)Mrows * (Gn * Kn * 2) * sizeof(float));
    float*  mlog    = (float*) alloc((size_t)Mrows * (Gn * Kn) * sizeof(float));
    bf16_t* dcn_bf  = (bf16_t*)alloc(actB);
    float*  y_f     = (float*) alloc(actF);
    float*  x2_f    = (float*) alloc(actF);
    bf16_t* x2_bf   = (bf16_t*)alloc(actB);
    bf16_t* h_bf    = (bf16_t*)alloc(actB);
    float*  m_f     = (float*) alloc(actF);
    bf16_t* w_in_t  = (bf16_t*)alloc((size_t)Cn * Cn * sizeof(bf16_t));
    bf16_t* w_off_t = (bf16_t*)alloc((size_t)(Gn * Kn * 2) * Cn * sizeof(bf16_t));
    bf16_t* w_mask_t= (bf16_t*)alloc((size_t)(Gn * Kn) * Cn * sizeof(bf16_t));
    bf16_t* w_out_t = (bf16_t*)alloc((size_t)Cn * Cn * sizeof(bf16_t));
    bf16_t* w_fc1_t = (bf16_t*)alloc((size_t)Cn * Cn * sizeof(bf16_t));
    bf16_t* w_fc2_t = (bf16_t*)alloc((size_t)Cn * Cn * sizeof(bf16_t));

    // weight transpose + bf16 convert
    auto wtrans = [&](const float* W, bf16_t* Wt, int K, int N) {
        int n = K * N;
        transpose_w_bf16<<<(n + 255) / 256, 256, 0, stream>>>(W, Wt, K, N);
    };
    wtrans(w_in,   w_in_t,   Cn, Cn);
    wtrans(w_off,  w_off_t,  Cn, Gn * Kn * 2);
    wtrans(w_mask, w_mask_t, Cn, Gn * Kn);
    wtrans(w_out,  w_out_t,  Cn, Cn);
    wtrans(w_fc1,  w_fc1_t,  Cn, Cn);
    wtrans(w_fc2,  w_fc2_t,  Cn, Cn);

    // x -> bf16 (A-side of value-projection GEMM)
    {
        int n = Mrows * Cn;
        cvt_bf16<<<(n + 255) / 256, 256, 0, stream>>>(x, x_bf, n);
    }

    // xp = x @ w_in + b_in
    launch_gemm<4, 2>(x_bf, w_in_t, b_in, xp_f, nullptr, Mrows, Cn, Cn, 0, stream);

    // x1 = GELU(LN(dwconv(x)))
    dwconv_ln_gelu<<<Mrows, Cn, 0, stream>>>(x, w_dw, b_dw, lndw_g, lndw_b, x1_f, x1_bf);

    // offset / mask-logit projections
    launch_gemm<4, 2>(x1_bf, w_off_t,  b_off,  offbuf, nullptr,
                      Mrows, Gn * Kn * 2, Cn, 0, stream);
    launch_gemm<4, 3>(x1_bf, w_mask_t, b_mask, mlog,   nullptr,
                      Mrows, Gn * Kn,     Cn, 0, stream);

    // DCNv3 core (softmax + bilinear gather)
    {
        int n = Mrows * Gn * GCn;   // B*HW*G*GC threads
        dcn_core<<<(n + 255) / 256, 256, 0, stream>>>(xp_f, offbuf, mlog, dcn_bf);
    }

    // y = dcn @ w_out + b_out
    launch_gemm<4, 2>(dcn_bf, w_out_t, b_out, y_f, nullptr, Mrows, Cn, Cn, 0, stream);

    // x2 = x + gamma1 * LN1(y)
    residual_ln<<<Mrows, Cn, 0, stream>>>(x, y_f, ln1_g, ln1_b, gamma1, x2_f, x2_bf);

    // h = GELU(x2 @ w_fc1 + b_fc1)   (bf16-only output, feeds next GEMM)
    launch_gemm<4, 2>(x2_bf, w_fc1_t, b_fc1, nullptr, h_bf, Mrows, Cn, Cn, 1, stream);

    // m = h @ w_fc2 + b_fc2
    launch_gemm<4, 2>(h_bf, w_fc2_t, b_fc2, m_f, nullptr, Mrows, Cn, Cn, 0, stream);

    // out = x2 + gamma2 * LN2(m)
    residual_ln<<<Mrows, Cn, 0, stream>>>(x2_f, m_f, ln2_g, ln2_b, gamma2,
                                          (float*)d_out, nullptr);
}
